// VineRNNModel_11991548690512
// MI455X (gfx1250) — matrix operand order repaired and checked by reading, not compiled, not run on hardware
//
#include <hip/hip_runtime.h>
#include <hip/hip_bf16.h>

// ---- problem dims ----
#define B   32
#define T   1024
#define H   512
#define E   256
#define V0  2048
#define V1  512
#define K2E 512      // 2*E, K of the gi GEMM; also H = K of gh/decoder GEMMs
#define N3H 1536     // 3*H

typedef __bf16 bf16;
typedef __attribute__((ext_vector_type(16))) __bf16 v16bf;
typedef __attribute__((ext_vector_type(8)))  __bf16 v8bf;
typedef __attribute__((ext_vector_type(2)))  __bf16 v2bf;
typedef __attribute__((ext_vector_type(8)))  float  v8f;

union FragU { v16bf v; v8bf h[2]; };

// A fragment (16x32 bf16, M x K), rows strided by ld elements.
// lane<16: M=lane,   K = {0..7, 16..23}
// lane>=16:M=lane-16,K = {8..15, 24..31}
__device__ __forceinline__ v16bf load_a_bf16(const bf16* base, int ld, int lane, int k0) {
  const bf16* p = base + (size_t)(lane & 15) * ld + k0 + ((lane >> 4) << 3);
  FragU u;
  u.h[0] = *(const v8bf*)(p);
  u.h[1] = *(const v8bf*)(p + 16);
  return u.v;
}

// B fragment (32x16 bf16, K x N) sourced from W stored [N, K] row-major
// (i.e. B = W^T, each lane reads two 8-K runs of one column n).
// lane<16: N=lane, K={0..7,16..23} ; lane>=16: N=lane-16, K={8..15,24..31}
__device__ __forceinline__ v16bf load_b_nk(const bf16* W, int ld, int n0, int lane, int k0) {
  const bf16* p = W + (size_t)(n0 + (lane & 15)) * ld + k0 + ((lane >> 4) << 3);
  FragU u;
  u.h[0] = *(const v8bf*)(p);
  u.h[1] = *(const v8bf*)(p + 16);
  return u.v;
}

__device__ __forceinline__ v8f wmma_bf16(v16bf a, v16bf b, v8f c) {
  return __builtin_amdgcn_wmma_f32_16x16x32_bf16(false, a, false, b, (short)0, c, false, false);
}

// ---------------- prep kernels ----------------
__global__ void f32_to_bf16_kernel(const float* __restrict__ src, bf16* __restrict__ dst, int n) {
  int i = blockIdx.x * 256 + threadIdx.x;
  if (i < n) dst[i] = (bf16)src[i];
}

__global__ void copy_f32_kernel(const float* __restrict__ src, float* __restrict__ dst, int n) {
  int i = blockIdx.x * 256 + threadIdx.x;
  if (i < n) dst[i] = src[i];
}

// ---------------- embedding gather -> X[t*B+b][512] bf16 ----------------
__global__ __launch_bounds__(128) void embed_kernel(
    const int* __restrict__ tok0, const int* __restrict__ tok1,
    const float* __restrict__ emb0, const float* __restrict__ emb1,
    bf16* __restrict__ X) {
  int bi = blockIdx.x;
  int t = bi >> 5, b = bi & 31;
  int i = threadIdx.x;                  // handles elements 2i, 2i+1
  int k0 = tok0[b * T + t];
  int k1 = tok1[b * T + t];
  const float* e0 = emb0 + (size_t)k0 * E + 2 * i;
  const float* e1 = emb1 + (size_t)k1 * E + 2 * i;
  bf16* xr = X + (size_t)(t * B + b) * K2E;
  v2bf p0; p0[0] = (bf16)e0[0]; p0[1] = (bf16)e0[1];
  v2bf p1; p1[0] = (bf16)e1[0]; p1[1] = (bf16)e1[1];
  *(v2bf*)(xr + 2 * i)     = p0;
  *(v2bf*)(xr + E + 2 * i) = p1;
}

// ---------------- gi = X @ W_ih^T + b_ih  (M=32768, K=512, N=1536) ----------------
// grid: (M/16) * 3 ; block 256 (8 waves); wave computes 16x64 strip
__global__ __launch_bounds__(256) void gemm_gi_kernel(
    const bf16* __restrict__ X, const bf16* __restrict__ Wih,
    const float* __restrict__ bih, float* __restrict__ gi) {
  int mt = blockIdx.x / 3;
  int nb = blockIdx.x % 3;
  int wave = threadIdx.x >> 5, lane = threadIdx.x & 31;
  int n0 = nb * 512 + wave * 64;
  const bf16* Abase = X + (size_t)(mt * 16) * K2E;
  v8f acc[4] = {};
  for (int k0 = 0; k0 < K2E; k0 += 32) {
    v16bf a = load_a_bf16(Abase, K2E, lane, k0);
#pragma unroll
    for (int j = 0; j < 4; ++j) {
      v16bf bfr = load_b_nk(Wih, K2E, n0 + j * 16, lane, k0);
      acc[j] = wmma_bf16(a, bfr, acc[j]);
    }
  }
  int rbase = mt * 16 + ((lane >> 4) << 3);
#pragma unroll
  for (int j = 0; j < 4; ++j) {
    int col = n0 + j * 16 + (lane & 15);
    float bv = bih[col];
#pragma unroll
    for (int r = 0; r < 8; ++r)   // gi is streaming (192MB, read-once): keep out of L2
      __builtin_nontemporal_store(acc[j][r] + bv, &gi[(size_t)(rbase + r) * N3H + col]);
  }
}

// ---------------- sequential GRU: single persistent workgroup (32 waves) ----------------
__global__ __launch_bounds__(1024) void gru_recurrent_kernel(
    const int* __restrict__ cond,     // [B, T]
    const float* __restrict__ gi,     // [T*B, 1536]
    const bf16* __restrict__ Whh,     // [1536, 512] bf16 (original [3H, H] layout)
    const float* __restrict__ bhh,    // [1536]
    float* __restrict__ buf,          // [T+1, B, H]  f32 state buffer
    float* __restrict__ gh,           // [B, 1536]    f32 scratch
    bf16* __restrict__ Hbf)           // [T+1, B, H]  bf16 mirror for decoders
{
  __shared__ __align__(16) bf16 hprev[B * H];    // 32 KB
  int tid  = threadIdx.x;
  int wave = tid >> 5, lane = tid & 31;
  int gb = tid >> 5;             // sample index for gather/gating
  int gj = (tid & 31) << 4;      // 16 consecutive features per thread
  int mt = wave & 1;             // this wave's M-tile (B=32 -> 2 tiles)
  int ntBase = (wave >> 1) * 6;  // 6 N-tiles per wave (96 total)

  for (int t = 0; t < T; ++t) {
    // 1) gather conditioned state, convert to bf16 into LDS
    {
      int c = cond[gb * T + t];
      const float* hsrc = buf + ((size_t)c * B + gb) * H + gj;
      bf16* hdst = hprev + gb * H + gj;
#pragma unroll
      for (int i = 0; i < 16; ++i) hdst[i] = (bf16)hsrc[i];
    }
    __syncthreads();

    // 2) gh = hprev @ Whh^T + bhh   (M=32, K=512, N=1536), WMMA from LDS
    {
      v8f acc[6] = {};
      const bf16* Abase = hprev + mt * 16 * H;
      for (int k0 = 0; k0 < H; k0 += 32) {
        v16bf a = load_a_bf16(Abase, H, lane, k0);
#pragma unroll
        for (int q = 0; q < 6; ++q) {
          v16bf bfr = load_b_nk(Whh, H, (ntBase + q) * 16, lane, k0);
          acc[q] = wmma_bf16(a, bfr, acc[q]);
        }
      }
      int rbase = mt * 16 + ((lane >> 4) << 3);
#pragma unroll
      for (int q = 0; q < 6; ++q) {
        int col = (ntBase + q) * 16 + (lane & 15);
        float bv = bhh[col];
#pragma unroll
        for (int r = 0; r < 8; ++r)
          gh[(size_t)(rbase + r) * N3H + col] = acc[q][r] + bv;
      }
    }
    __syncthreads();

    // 3) gating, write new state buf[t+1] (f32) + Hbf[t+1] (bf16 for decoders)
    {
      int c = cond[gb * T + t];
      const float* gip = gi  + ((size_t)t * B + gb) * N3H;
      const float* ghp = gh  + (size_t)gb * N3H;
      const float* hp  = buf + ((size_t)c * B + gb) * H;
      float* hout      = buf + ((size_t)(t + 1) * B + gb) * H;
      bf16*  hbfout    = Hbf + ((size_t)(t + 1) * B + gb) * H + gj;
      FragU hb;
#pragma unroll
      for (int i = 0; i < 16; ++i) {
        int j = gj + i;
        float ir = __builtin_nontemporal_load(&gip[j]);
        float iz = __builtin_nontemporal_load(&gip[512 + j]);
        float inn = __builtin_nontemporal_load(&gip[1024 + j]);
        float hr = ghp[j], hz = ghp[512 + j], hn  = ghp[1024 + j];
        float r = 1.f / (1.f + __expf(-(ir + hr)));
        float z = 1.f / (1.f + __expf(-(iz + hz)));
        float n = tanhf(inn + r * hn);
        float hv = (1.f - z) * n + z * hp[j];
        hout[j] = hv;
        hb.v[i] = (bf16)hv;
      }
      *(v8bf*)(hbfout)     = hb.h[0];   // 16B stores, gj*2 is 32B aligned
      *(v8bf*)(hbfout + 8) = hb.h[1];
    }
    __syncthreads();
  }
}

// ---------------- decoders: out[b*T+t][v] = Hbf[t+1][b] . W[v] + bias[v] ----------------
// grid: (M/16) * nbCount ; wave computes 16x64; A rows strided by B*H in Hbf
__global__ __launch_bounds__(256) void gemm_dec_kernel(
    const bf16* __restrict__ Hbf, const bf16* __restrict__ W,
    const float* __restrict__ bias, float* __restrict__ out,
    int V, int nbCount) {
  int mt = blockIdx.x / nbCount;
  int nb = blockIdx.x % nbCount;
  int wave = threadIdx.x >> 5, lane = threadIdx.x & 31;
  int n0 = nb * 512 + wave * 64;
  int ro = mt * 16;            // output rows ro..ro+15 (row = b*T + t)
  int bb = ro >> 10;           // batch (tiles never straddle b since 16 | 1024)
  int t0 = ro & 1023;
  // A tile base: row m -> Hbf[((t0+m)+1)*B + bb][.] ; row stride = B*H elements
  const bf16* Abase = Hbf + ((size_t)(t0 + 1) * B + bb) * H;

  v8f acc[4] = {};
  for (int k0 = 0; k0 < H; k0 += 32) {
    v16bf a = load_a_bf16(Abase, B * H, lane, k0);
#pragma unroll
    for (int j = 0; j < 4; ++j) {
      v16bf bfr = load_b_nk(W, H, n0 + j * 16, lane, k0);
      acc[j] = wmma_bf16(a, bfr, acc[j]);
    }
  }
  int rbase = ro + ((lane >> 4) << 3);
#pragma unroll
  for (int j = 0; j < 4; ++j) {
    int col = n0 + j * 16 + (lane & 15);
    float bv = bias[col];
#pragma unroll
    for (int r = 0; r < 8; ++r)   // outputs are write-once streams (336MB): NT stores
      __builtin_nontemporal_store(acc[j][r] + bv, &out[(size_t)(rbase + r) * V + col]);
  }
}

// ---------------- launch ----------------
extern "C" void kernel_launch(void* const* d_in, const int* in_sizes, int n_in,
                              void* d_out, int out_size, void* d_ws, size_t ws_size,
                              hipStream_t stream) {
  const int*   tokens0 = (const int*)d_in[0];
  const int*   tokens1 = (const int*)d_in[1];
  const int*   cond    = (const int*)d_in[2];
  const float* hidden  = (const float*)d_in[3];
  const float* emb0    = (const float*)d_in[4];
  const float* emb1    = (const float*)d_in[5];
  const float* W_ih    = (const float*)d_in[6];
  const float* W_hh    = (const float*)d_in[7];
  const float* b_ih    = (const float*)d_in[8];
  const float* b_hh    = (const float*)d_in[9];
  const float* dec0_W  = (const float*)d_in[10];
  const float* dec0_b  = (const float*)d_in[11];
  const float* dec1_W  = (const float*)d_in[12];
  const float* dec1_b  = (const float*)d_in[13];

  // workspace layout (bytes)
  char* ws = (char*)d_ws;
  bf16*  Xbf   = (bf16*)(ws);                               // 32 MB    [T*B, 512]
  bf16*  WihBf = (bf16*)(ws + 33554432);                    // 1.5 MB   [1536, 512]
  bf16*  WhhBf = (bf16*)(ws + 35127296);                    // 1.5 MB   [1536, 512]
  bf16*  W0Bf  = (bf16*)(ws + 36700160);                    // 2 MB     [2048, 512]
  bf16*  W1Bf  = (bf16*)(ws + 38797312);                    // 0.5 MB   [512, 512]
  float* gh    = (float*)(ws + 39321600);                   // 192 KB   [32, 1536]
  float* gi    = (float*)(ws + 39518208);                   // 192 MB   [T*B, 1536]
  float* buf   = (float*)(ws + 240844800);                  // 64 MB    [T+1, B, H] f32
  bf16*  Hbf   = (bf16*)(ws + 308019200);                   // 32 MB    [T+1, B, H] bf16

  float* out0 = (float*)d_out;                               // [B, T, V0]
  float* out1 = out0 + (size_t)B * T * V0;                   // [B, T, V1]
  float* outH = out1 + (size_t)B * T * V1;                   // [B, H]

  // weight conversions + initial state
  f32_to_bf16_kernel<<<(N3H * K2E + 255) / 256, 256, 0, stream>>>(W_ih, WihBf, N3H * K2E);
  f32_to_bf16_kernel<<<(N3H * H   + 255) / 256, 256, 0, stream>>>(W_hh, WhhBf, N3H * H);
  f32_to_bf16_kernel<<<(V0  * H   + 255) / 256, 256, 0, stream>>>(dec0_W, W0Bf, V0 * H);
  f32_to_bf16_kernel<<<(V1  * H   + 255) / 256, 256, 0, stream>>>(dec1_W, W1Bf, V1 * H);
  copy_f32_kernel<<<(B * H + 255) / 256, 256, 0, stream>>>(hidden, buf, B * H);
  f32_to_bf16_kernel<<<(B * H + 255) / 256, 256, 0, stream>>>(hidden, Hbf, B * H);

  // embeddings -> X (bf16, time-major)
  embed_kernel<<<B * T, 128, 0, stream>>>(tokens0, tokens1, emb0, emb1, Xbf);

  // gi = X @ W_ih^T + b_ih (parallel over all timesteps)
  gemm_gi_kernel<<<(T * B / 16) * 3, 256, 0, stream>>>(Xbf, WihBf, b_ih, gi);

  // sequential GRU on one persistent workgroup
  gru_recurrent_kernel<<<1, 1024, 0, stream>>>(cond, gi, WhhBf, b_hh, buf, gh, Hbf);

  // decoders (read bf16 state mirror)
  gemm_dec_kernel<<<(T * B / 16) * 4, 256, 0, stream>>>(Hbf, W0Bf, dec0_b, out0, V0, 4);
  gemm_dec_kernel<<<(T * B / 16) * 1, 256, 0, stream>>>(Hbf, W1Bf, dec1_b, out1, V1, 1);

  // final hidden = buf[T]
  copy_f32_kernel<<<(B * H + 255) / 256, 256, 0, stream>>>(buf + (size_t)T * B * H, outH, B * H);
}
